// CasualAttention_29308856827927
// MI455X (gfx1250) — matrix operand order repaired and checked by reading, not compiled, not run on hardware
//
#include <hip/hip_runtime.h>
#include <hip/hip_bf16.h>
#include <stdint.h>

// ---------------------------------------------------------------------------
// Causal attention, B=4, S=2048, D=1024, fp32 in/out, bf16 WMMA internally.
// Workspace: [0,16MB) Q bf16 | [16,32MB) K bf16 | [32,48MB) V bf16 |
//            [48,112MB) scores/probs fp32 (softmax in place). Fits in L2.
// GEMM scheme: block = 256 threads = 8 waves; block tile 128(M)x256(N),
// K-step 32; wave tile 64x64 = 4x4 WMMA accumulators (16 wmma : 16 ds_load).
// Score kernel additionally uses async global->LDS copies when available.
// ---------------------------------------------------------------------------

#define B_  4
#define S_  2048
#define D_  1024
#define LDSS 40          // LDS row stride in bf16 (32 + 8 pad; 80B, 16B aligned)

typedef __attribute__((ext_vector_type(16))) __bf16 v16bf;
typedef __attribute__((ext_vector_type(8)))  float  v8f;
typedef __attribute__((ext_vector_type(4)))  int    v4i;

struct __align__(16) FragBits { uint4 lo, hi; };

// ---- async global->LDS 16B copy (CDNA5), with portable fallback ------------
#if defined(__HIP_DEVICE_COMPILE__) && \
    __has_builtin(__builtin_amdgcn_global_load_async_to_lds_b128)
#define HAVE_ASYNC_LDS 1
#endif

__device__ __forceinline__ void async_cp16(void* lds_dst, const void* gsrc) {
#if defined(HAVE_ASYNC_LDS)
    __builtin_amdgcn_global_load_async_to_lds_b128(
        (v4i*)(void*)gsrc,
        (__attribute__((address_space(3))) v4i*)lds_dst,
        /*offset=*/0, /*cpol=*/0);
#else
    *reinterpret_cast<uint4*>(lds_dst) = *reinterpret_cast<const uint4*>(gsrc);
#endif
}

__device__ __forceinline__ void async_wait_all() {
#if defined(HAVE_ASYNC_LDS)
#if __has_builtin(__builtin_amdgcn_s_wait_asynccnt)
    __builtin_amdgcn_s_wait_asynccnt(0);
#else
    asm volatile("s_wait_asynccnt 0x0" ::: "memory");
#endif
#endif
}

// ---- fragment loaders matching CDNA5 16-bit WMMA VGPR layouts --------------
// A (16x32 MxK) from As[m][k]: lanes0-15 row=lane (K 0..7 / 16..23),
// lanes16-31 row=lane-16 (K 8..15 / 24..31)
__device__ inline v16bf load_frag_a(const __bf16* As, int mBase, int lane) {
    const int row = mBase + (lane & 15);
    const int kh  = (lane >> 4) << 3;                 // 0 or 8
    const uint4* p = reinterpret_cast<const uint4*>(As + row * LDSS + kh);
    FragBits f;
    f.lo = p[0];                                      // K = kh .. kh+7
    f.hi = p[2];                                      // K = kh+16 .. kh+23
    return __builtin_bit_cast(v16bf, f);
}
// B (32x16 KxN) from Bs[n][k]: lanes0-15 col=lane K 0..7 / 16..23; lanes16-31 K 8..15 / 24..31
__device__ inline v16bf load_frag_b(const __bf16* Bs, int nBase, int lane) {
    const int row = nBase + (lane & 15);
    const int kh  = (lane >> 4) << 3;                 // 0 or 8
    const uint4* p = reinterpret_cast<const uint4*>(Bs + row * LDSS + kh);
    FragBits f;
    f.lo = p[0];
    f.hi = p[2];
    return __builtin_bit_cast(v16bf, f);
}

// shared inner product step: 4x4 wmma on staged tiles
#define WAVE_COORDS()                                    \
    const int tid  = threadIdx.x;                        \
    const int lane = tid & 31;                           \
    const int wv   = tid >> 5;                           \
    const int wm   = (wv >> 2) * 64;  /* 2 wave rows */  \
    const int wn   = (wv & 3) * 64    /* 4 wave cols */

#define MMA_STEP(As, Bs)                                                      \
    do {                                                                      \
        v16bf a[4], b[4];                                                     \
        _Pragma("unroll")                                                     \
        for (int i = 0; i < 4; ++i) a[i] = load_frag_a(As, wm + i * 16, lane);\
        _Pragma("unroll")                                                     \
        for (int j = 0; j < 4; ++j) b[j] = load_frag_b(Bs, wn + j * 16, lane);\
        _Pragma("unroll")                                                     \
        for (int i = 0; i < 4; ++i)                                           \
            _Pragma("unroll")                                                 \
            for (int j = 0; j < 4; ++j)                                       \
                acc[i][j] = __builtin_amdgcn_wmma_f32_16x16x32_bf16(          \
                    false, a[i], false, b[j], (short)0, acc[i][j],            \
                    false, false);                                            \
    } while (0)

// ===========================================================================
// Kernel 1: projections. X[8192,1024] fp32 x W[1024,1024] fp32 -> bf16.
// grid = (N/256=4, M/128=64, 3)
// ===========================================================================
__global__ __launch_bounds__(256) void proj_kernel(
    const float* __restrict__ X,
    const float* __restrict__ Wq, const float* __restrict__ Wk,
    const float* __restrict__ Wv,
    __bf16* __restrict__ Q, __bf16* __restrict__ K, __bf16* __restrict__ V)
{
    const float* W; __bf16* Out;
    if      (blockIdx.z == 0) { W = Wq; Out = Q; }
    else if (blockIdx.z == 1) { W = Wk; Out = K; }
    else                      { W = Wv; Out = V; }

    const int nBase = blockIdx.x * 256;
    const int mBase = blockIdx.y * 128;

    __shared__ __bf16 As[128 * LDSS];
    __shared__ __bf16 Bs[256 * LDSS];

    WAVE_COORDS();
    v8f acc[4][4] = {};

    for (int k0 = 0; k0 < D_; k0 += 32) {
        // stage A: 128x32 fp32 -> bf16, packed 4-wide (8B) LDS stores
        {
            const int r = tid >> 3;              // 0..31
            const int c = (tid & 7) * 4;         // 0..28
            #pragma unroll
            for (int pass = 0; pass < 4; ++pass) {
                const int row = r + pass * 32;
                const float4 f = *reinterpret_cast<const float4*>(
                    X + (size_t)(mBase + row) * D_ + k0 + c);
                __bf16 h[4] = {(__bf16)f.x, (__bf16)f.y, (__bf16)f.z, (__bf16)f.w};
                *reinterpret_cast<uint2*>(As + row * LDSS + c) =
                    *reinterpret_cast<uint2*>(h);
            }
        }
        // stage B transposed: Bs[n][k] = W[k0+k][nBase+n].
        // Each thread owns column n, packs K-pairs -> b32 LDS stores.
        {
            const int n = tid;                    // 0..255
            #pragma unroll
            for (int kp = 0; kp < 16; ++kp) {
                const float f0 = W[(size_t)(k0 + 2 * kp) * D_ + nBase + n];
                const float f1 = W[(size_t)(k0 + 2 * kp + 1) * D_ + nBase + n];
                __bf16 h[2] = {(__bf16)f0, (__bf16)f1};
                *reinterpret_cast<uint32_t*>(Bs + n * LDSS + 2 * kp) =
                    *reinterpret_cast<uint32_t*>(h);
            }
        }
        __syncthreads();
        MMA_STEP(As, Bs);
        __syncthreads();
    }

    const int col = lane & 15;
    const int rb  = (lane >> 4) * 8;
    #pragma unroll
    for (int i = 0; i < 4; ++i)
        #pragma unroll
        for (int j = 0; j < 4; ++j) {
            const int m0 = mBase + wm + i * 16 + rb;
            const int n0 = nBase + wn + j * 16 + col;
            #pragma unroll
            for (int v = 0; v < 8; ++v)
                Out[(size_t)(m0 + v) * D_ + n0] = (__bf16)acc[i][j][v];
        }
}

// ===========================================================================
// Kernel 2: scores = Q.K^T per batch; tiles fully above diagonal skipped.
// Staging uses async global->LDS (pure bf16 copies). grid = (8, 16, 4)
// ===========================================================================
__global__ __launch_bounds__(256) void score_kernel(
    const __bf16* __restrict__ Q, const __bf16* __restrict__ Kmat,
    float* __restrict__ Sc)
{
    const int kBase = blockIdx.x * 256;
    const int qBase = blockIdx.y * 128;
    if (kBase >= qBase + 128) return;              // fully masked tile
    const int bb = blockIdx.z;

    const __bf16* Qb = Q    + (size_t)bb * S_ * D_;
    const __bf16* Kb = Kmat + (size_t)bb * S_ * D_;
    float*        Sb = Sc   + (size_t)bb * S_ * S_;

    __shared__ __bf16 As[128 * LDSS];
    __shared__ __bf16 Bs[256 * LDSS];

    WAVE_COORDS();
    v8f acc[4][4] = {};

    for (int k0 = 0; k0 < D_; k0 += 32) {
        // stage A = Q tile rows: async 16B copies
        {
            const int r = tid >> 2;              // 0..63
            const int c = (tid & 3) * 8;
            #pragma unroll
            for (int pass = 0; pass < 2; ++pass) {
                const int row = r + pass * 64;
                async_cp16(As + row * LDSS + c,
                           Qb + (size_t)(qBase + row) * D_ + k0 + c);
            }
        }
        // stage B: Bs[n][k] = K[key=kBase+n][k0+k] -> straight row copy
        {
            const int r = tid >> 2;              // 0..63
            const int c = (tid & 3) * 8;
            #pragma unroll
            for (int pass = 0; pass < 4; ++pass) {
                const int row = r + pass * 64;
                async_cp16(Bs + row * LDSS + c,
                           Kb + (size_t)(kBase + row) * D_ + k0 + c);
            }
        }
        async_wait_all();
        __syncthreads();
        MMA_STEP(As, Bs);
        __syncthreads();
    }

    const int col = lane & 15;
    const int rb  = (lane >> 4) * 8;
    #pragma unroll
    for (int i = 0; i < 4; ++i)
        #pragma unroll
        for (int j = 0; j < 4; ++j) {
            const int m0 = qBase + wm + i * 16 + rb;
            const int n0 = kBase + wn + j * 16 + col;
            #pragma unroll
            for (int v = 0; v < 8; ++v)
                Sb[(size_t)(m0 + v) * S_ + n0] = acc[i][j][v];
        }
}

// ===========================================================================
// Kernel 3: in-place causal scaled softmax per score row. grid = B*S.
// ===========================================================================
__global__ __launch_bounds__(256) void softmax_kernel(float* __restrict__ Sc)
{
    const int row = blockIdx.x;                    // 0 .. B*S-1
    const int i   = row & (S_ - 1);
    float* s = Sc + (size_t)row * S_;
    const int valid = i + 1;
    const float scale = 0.03125f;                  // 1/sqrt(1024)

    __shared__ float red[256];
    const int tid = threadIdx.x;

    float vbuf[S_ / 256];                          // scaled scores cache (<=8)
    int cnt = 0;
    float m = -__builtin_inff();
    for (int j = tid; j < valid; j += 256) {
        const float v = s[j] * scale;
        vbuf[cnt++] = v;
        m = fmaxf(m, v);
    }
    red[tid] = m; __syncthreads();
    #pragma unroll
    for (int off = 128; off > 0; off >>= 1) {
        if (tid < off) red[tid] = fmaxf(red[tid], red[tid + off]);
        __syncthreads();
    }
    m = red[0];
    __syncthreads();

    float sum = 0.f;
    for (int t = 0; t < cnt; ++t) {
        vbuf[t] = __expf(vbuf[t] - m);
        sum += vbuf[t];
    }
    red[tid] = sum; __syncthreads();
    #pragma unroll
    for (int off = 128; off > 0; off >>= 1) {
        if (tid < off) red[tid] += red[tid + off];
        __syncthreads();
    }
    const float inv = 1.0f / red[0];

    {
        int t = 0;
        for (int j = tid; j < valid; j += 256) s[j] = vbuf[t++] * inv;
    }
    for (int j = valid + tid; j < S_; j += 256) s[j] = 0.f;
}

// ===========================================================================
// Kernel 4: out = P.V per batch; K-loop truncated at causal limit.
// grid = (D/256=4, queries/128=16, B=4)
// ===========================================================================
__global__ __launch_bounds__(256) void pv_kernel(
    const float* __restrict__ P, const __bf16* __restrict__ V,
    float* __restrict__ Out)
{
    const int nBase = blockIdx.x * 256;
    const int qBase = blockIdx.y * 128;
    const int bb    = blockIdx.z;

    const float*  Pb = P   + (size_t)bb * S_ * S_;
    const __bf16* Vb = V   + (size_t)bb * S_ * D_;
    float*        Ob = Out + (size_t)bb * S_ * D_;

    __shared__ __bf16 As[128 * LDSS];
    __shared__ __bf16 Bs[256 * LDSS];

    WAVE_COORDS();
    v8f acc[4][4] = {};

    const int kmax = qBase + 128;                  // P[q][k]==0 for k > q

    for (int k0 = 0; k0 < kmax; k0 += 32) {
        // stage A: probabilities fp32 -> bf16, packed 8B LDS stores
        {
            const int r = tid >> 3;
            const int c = (tid & 7) * 4;
            #pragma unroll
            for (int pass = 0; pass < 4; ++pass) {
                const int row = r + pass * 32;
                const float4 f = *reinterpret_cast<const float4*>(
                    Pb + (size_t)(qBase + row) * S_ + k0 + c);
                __bf16 h[4] = {(__bf16)f.x, (__bf16)f.y, (__bf16)f.z, (__bf16)f.w};
                *reinterpret_cast<uint2*>(As + row * LDSS + c) =
                    *reinterpret_cast<uint2*>(h);
            }
        }
        // stage B transposed: Bs[n][k] = V[k0+k][nBase+n].
        // Read K-pair rows as uint4 (8 bf16), pack pairs -> b32 LDS stores.
        {
            const int n   = (tid & 31) * 8;       // 0..248
            const int kp0 = tid >> 5;             // 0..7
            #pragma unroll
            for (int t = 0; t < 2; ++t) {
                const int k = 2 * (kp0 + t * 8);  // even, 0..30
                uint4 r0 = *reinterpret_cast<const uint4*>(
                    Vb + (size_t)(k0 + k) * D_ + nBase + n);
                uint4 r1 = *reinterpret_cast<const uint4*>(
                    Vb + (size_t)(k0 + k + 1) * D_ + nBase + n);
                const __bf16* e0 = reinterpret_cast<const __bf16*>(&r0);
                const __bf16* e1 = reinterpret_cast<const __bf16*>(&r1);
                #pragma unroll
                for (int i = 0; i < 8; ++i) {
                    __bf16 h[2] = {e0[i], e1[i]};
                    *reinterpret_cast<uint32_t*>(Bs + (n + i) * LDSS + k) =
                        *reinterpret_cast<uint32_t*>(h);
                }
            }
        }
        __syncthreads();
        MMA_STEP(As, Bs);
        __syncthreads();
    }

    const int col = lane & 15;
    const int rb  = (lane >> 4) * 8;
    #pragma unroll
    for (int i = 0; i < 4; ++i)
        #pragma unroll
        for (int j = 0; j < 4; ++j) {
            const int m0 = qBase + wm + i * 16 + rb;
            const int n0 = nBase + wn + j * 16 + col;
            #pragma unroll
            for (int v = 0; v < 8; ++v)
                Ob[(size_t)(m0 + v) * D_ + n0] = acc[i][j][v];
        }
}

// ===========================================================================
extern "C" void kernel_launch(void* const* d_in, const int* in_sizes, int n_in,
                              void* d_out, int out_size, void* d_ws, size_t ws_size,
                              hipStream_t stream)
{
    (void)in_sizes; (void)n_in; (void)out_size; (void)ws_size;

    const float* x  = (const float*)d_in[0];   // [4,2048,1024]
    const float* Wq = (const float*)d_in[1];   // [1024,1024]
    const float* Wk = (const float*)d_in[2];
    const float* Wv = (const float*)d_in[3];
    float* out = (float*)d_out;                // [4,2048,1024]

    char* ws = (char*)d_ws;
    const size_t qkvElems = (size_t)B_ * S_ * D_;
    __bf16* Q  = (__bf16*)ws;
    __bf16* K  = Q + qkvElems;
    __bf16* V  = K + qkvElems;
    float*  Sc = (float*)(ws + 3 * qkvElems * sizeof(__bf16));

    proj_kernel<<<dim3(D_ / 256, (B_ * S_) / 128, 3), 256, 0, stream>>>(
        x, Wq, Wk, Wv, Q, K, V);

    score_kernel<<<dim3(S_ / 256, S_ / 128, B_), 256, 0, stream>>>(Q, K, Sc);

    softmax_kernel<<<dim3(B_ * S_), 256, 0, stream>>>(Sc);

    pv_kernel<<<dim3(D_ / 256, S_ / 128, B_), 256, 0, stream>>>(Sc, V, out);
}
